// D_Mixed_5b_88201448391307
// MI455X (gfx1250) — matrix operand order repaired and checked by reading, not compiled, not run on hardware
//
#include <hip/hip_runtime.h>
#include <hip/hip_bf16.h>
#include <cstddef>

// Problem geometry (fixed by the reference).
#define TB 8
#define TT 8
#define THh 14
#define TWw 14
#define HWX (THh * TWw)          // 196
#define THW (TT * HWX)           // 1568
#define NPOS (TB * THW)          // 12544 = 784 * 16 = 49 * 256

#define SLD 34                   // padded LDS stride for the S tile (floats)

typedef __attribute__((ext_vector_type(2)))  float     v2f;
typedef __attribute__((ext_vector_type(8)))  float     v8f;
typedef __attribute__((ext_vector_type(16))) _Float16  v16h;

#ifdef __has_builtin
#if __has_builtin(__builtin_amdgcn_wmma_f32_16x16x4_f32)
#define HAVE_WMMA_F32_4 1
#endif
#endif

// ---------------------------------------------------------------------------
// One 32-deep K-chunk of a 16x16 output tile.
// Wrow: &W[mbase][kd0]  (row-major, leading dim = ldw = Cin*K, even)
// St:   LDS S-tile, layout St[pcol*SLD + j], j = K index within chunk (0..31)
// Exact fp32 via V_WMMA_F32_16X16X4_F32 (8 steps of K=4); A/B frags are
// contiguous K-pairs -> single b64 loads.
// ---------------------------------------------------------------------------
__device__ __forceinline__ v8f mma_chunk(const float* __restrict__ Wrow, int ldw,
                                         const float* __restrict__ St,
                                         int lane, v8f c) {
  const int n    = lane & 15;
  const int half = lane >> 4;
  const float* wr = Wrow + (size_t)n * ldw;
  const float* sr = St + n * SLD;
#ifdef HAVE_WMMA_F32_4
  #pragma unroll
  for (int ks = 0; ks < 8; ++ks) {
    // A 16x4 f32 layout: VGPR0 = K0 | K2, VGPR1 = K1 | K3 (M = lane&15)
    // B mirrored (N = lane&15). Both pairs contiguous & 8B aligned.
    const int kb = ks * 4 + (half ? 2 : 0);
    v2f a = *(const v2f*)(wr + kb);
    v2f b = *(const v2f*)(sr + kb);
    c = __builtin_amdgcn_wmma_f32_16x16x4_f32(false, a, false, b,
                                              (short)0, c, false, false);
  }
#else
  // Split-f16 (hi+lo) fallback: 3x V_WMMA_F32_16X16X32_F16, ~fp32 accuracy.
  v16h ah, al, bh, bl;
  #pragma unroll
  for (int v = 0; v < 8; ++v) {
    const int kA = ((v < 4) ? (2 * v) : (8 + 2 * v)) + (half ? 8 : 0);
    const int kB = ((v < 4) ? (2 * v) : (8 + 2 * v)) + (half ? 8 : 0);
    float wa0 = wr[kA], wa1 = wr[kA + 1];
    float sb0 = sr[kB], sb1 = sr[kB + 1];
    _Float16 h0 = (_Float16)wa0, h1 = (_Float16)wa1;
    ah[2*v] = h0;  ah[2*v+1] = h1;
    al[2*v] = (_Float16)(wa0 - (float)h0);
    al[2*v+1] = (_Float16)(wa1 - (float)h1);
    _Float16 g0 = (_Float16)sb0, g1 = (_Float16)sb1;
    bh[2*v] = g0;  bh[2*v+1] = g1;
    bl[2*v] = (_Float16)(sb0 - (float)g0);
    bl[2*v+1] = (_Float16)(sb1 - (float)g1);
  }
  c = __builtin_amdgcn_wmma_f32_16x16x32_f16(false, ah, false, bh, (short)0, c, false, false);
  c = __builtin_amdgcn_wmma_f32_16x16x32_f16(false, ah, false, bl, (short)0, c, false, false);
  c = __builtin_amdgcn_wmma_f32_16x16x32_f16(false, al, false, bh, (short)0, c, false, false);
#endif
  return c;
}

// ---------------------------------------------------------------------------
// 1x1x1 offset conv (Cin = 832, 3 output channels). off[comp*NPOS + p].
// ---------------------------------------------------------------------------
__global__ __launch_bounds__(128)
void offc1(const float* __restrict__ Y, const float* __restrict__ woff,
           const float* __restrict__ boff, float* __restrict__ off) {
  const int p = blockIdx.x * 128 + threadIdx.x;     // grid = NPOS/128
  const int b = p / THW, s = p - b * THW;
  const float* base = Y + (size_t)b * 832 * THW + s;
  float a0 = boff[0], a1 = boff[1], a2 = boff[2];
  for (int c = 0; c < 832; ++c) {
    float xv = base[(size_t)c * THW];
    a0 += xv * woff[c];
    a1 += xv * woff[832 + c];
    a2 += xv * woff[2 * 832 + c];
  }
  off[p] = a0;
  off[NPOS + p] = a1;
  off[2 * NPOS + p] = a2;
}

// ---------------------------------------------------------------------------
// 3x3x3 offset conv, pad=1, 81 output channels. off[ch*NPOS + p].
// ---------------------------------------------------------------------------
__global__ __launch_bounds__(256)
void offc3(const float* __restrict__ Y, const float* __restrict__ woff,
           const float* __restrict__ boff, float* __restrict__ off, int Cin) {
  const int idx = blockIdx.x * 256 + threadIdx.x;   // grid = 81*NPOS/256
  const int ch = idx / NPOS, p = idx - ch * NPOS;
  const int b = p / THW, s = p - b * THW;
  const int t = s / HWX, r = s - t * HWX, h = r / TWw, w = r - h * TWw;
  const float* yb = Y + (size_t)b * Cin * THW;
  const float* wr = woff + (size_t)ch * Cin * 27;
  float acc = boff[ch];
  for (int kk = 0; kk < 27; ++kk) {
    const int tt = t + kk / 9 - 1;
    const int hh = h + (kk / 3) % 3 - 1;
    const int ww = w + kk % 3 - 1;
    if (tt < 0 || tt >= TT || hh < 0 || hh >= THh || ww < 0 || ww >= TWw) continue;
    const float* yp = yb + (tt * THh + hh) * TWw + ww;
    const float* wp = wr + kk;
    for (int c = 0; c < Cin; ++c)
      acc += yp[(size_t)c * THW] * wp[(size_t)c * 27];
  }
  off[(size_t)ch * NPOS + p] = acc;
}

// ---------------------------------------------------------------------------
// Fused deformable gather + WMMA GEMM.  pre[o*NPOS + p] = sum_kd W[o,kd]*S[kd,p]
// Block: 128 threads (4 waves), one block per 16-position tile.
// CIN/COUT are compile-time so accumulators stay in VGPRs (no scratch spill).
// ---------------------------------------------------------------------------
template <int K, int PAD, int CIN, int COUT>
__global__ __launch_bounds__(128)
void deform_gemm(const float* __restrict__ Y, const float* __restrict__ off,
                 const float* __restrict__ Wg, float* __restrict__ pre) {
  constexpr int Kdim = CIN * K;
  constexpr int nMt  = COUT / 16;      // 16x16 output tiles along Cout
  constexpr int MAXT = (nMt + 3) / 4;  // tiles per wave (upper bound)

  __shared__ __align__(16) float Stile[16 * SLD];
  __shared__ float cWgt[16 * K * 8];
  __shared__ int   cIdx[16 * K * 8];

  const int tid   = threadIdx.x;
  const int lane  = tid & 31;
  const int wv    = tid >> 5;
  const int pbase = blockIdx.x * 16;

  // Phase 0: trilinear weights + clamped indices for 16 positions x K taps.
  for (int e = tid; e < 16 * K; e += 128) {
    const int pc = e / K, kk = e - pc * K;
    const int p = pbase + pc;
    const int b = p / THW, s = p - b * THW;
    const int t = s / HWX, r = s - t * HWX, h = r / TWw, w = r - h * TWw;
    const int kt = (K == 27) ? kk / 9 : 0;
    const int kh = (K == 27) ? (kk / 3) % 3 : 0;
    const int kw = (K == 27) ? kk % 3 : 0;
    const float tc = (float)(t - PAD + kt) + off[(size_t)(kk * 3 + 0) * NPOS + p];
    const float hc = (float)(h - PAD + kh) + off[(size_t)(kk * 3 + 1) * NPOS + p];
    const float wc = (float)(w - PAD + kw) + off[(size_t)(kk * 3 + 2) * NPOS + p];
    const float t0 = floorf(tc), h0 = floorf(hc), w0 = floorf(wc);
    #pragma unroll
    for (int j = 0; j < 8; ++j) {
      const float ti = t0 + (float)((j >> 2) & 1);
      const float hi = h0 + (float)((j >> 1) & 1);
      const float wi = w0 + (float)(j & 1);
      float wgt = (1.f - fabsf(tc - ti)) * (1.f - fabsf(hc - hi)) * (1.f - fabsf(wc - wi));
      const bool valid = (ti >= 0.f) && (ti <= (float)(TT - 1)) &&
                         (hi >= 0.f) && (hi <= (float)(THh - 1)) &&
                         (wi >= 0.f) && (wi <= (float)(TWw - 1));
      const int it = (int)fminf(fmaxf(ti, 0.f), (float)(TT - 1));
      const int ih = (int)fminf(fmaxf(hi, 0.f), (float)(THh - 1));
      const int iw = (int)fminf(fmaxf(wi, 0.f), (float)(TWw - 1));
      cWgt[(pc * K + kk) * 8 + j] = valid ? wgt : 0.f;
      cIdx[(pc * K + kk) * 8 + j] = (it * THh + ih) * TWw + iw;
    }
  }
  __syncthreads();

  v8f acc[MAXT];
  #pragma unroll
  for (int i = 0; i < MAXT; ++i)
    #pragma unroll
    for (int j = 0; j < 8; ++j) acc[i][j] = 0.f;

  for (int kd0 = 0; kd0 < Kdim; kd0 += 32) {
    // Cooperative gather of the 32x16 S chunk into LDS (position-major).
    for (int i = tid; i < 512; i += 128) {
      const int j = i >> 4, pc = i & 15;
      const int kd = kd0 + j;
      const int c = kd / K, kk = kd - c * K;
      const int p = pbase + pc;
      const int b = p / THW;
      const float* base = Y + (size_t)(b * CIN + c) * THW;
      const float* wp = &cWgt[(pc * K + kk) * 8];
      const int*   ip = &cIdx[(pc * K + kk) * 8];
      float a = 0.f;
      #pragma unroll
      for (int q = 0; q < 8; ++q) a += wp[q] * base[ip[q]];
      Stile[pc * SLD + j] = a;
    }
    __syncthreads();

    #pragma unroll
    for (int ai = 0; ai < MAXT; ++ai) {
      const int mt = wv + 4 * ai;
      if (mt < nMt)
        acc[ai] = mma_chunk(Wg + (size_t)(mt * 16) * Kdim + kd0, Kdim,
                            Stile, lane, acc[ai]);
    }
    __syncthreads();
  }

  // Store pre-BN output. C/D layout: VGPR i -> M = i (lanes 0-15) / i+8 (16-31).
  const int n = lane & 15, half = lane >> 4;
  #pragma unroll
  for (int ai = 0; ai < MAXT; ++ai) {
    const int mt = wv + 4 * ai;
    if (mt < nMt) {
      #pragma unroll
      for (int i = 0; i < 8; ++i) {
        const int o = mt * 16 + i + (half << 3);
        pre[(size_t)o * NPOS + pbase + n] = acc[ai][i];
      }
    }
  }
}

// ---------------------------------------------------------------------------
// Deterministic per-channel sum / sum-of-squares (one block per channel).
// ---------------------------------------------------------------------------
__global__ __launch_bounds__(256)
void bnstats(const float* __restrict__ pre, float* __restrict__ bsum,
             float* __restrict__ bsq) {
  __shared__ float ls[256], lq[256];
  const int o = blockIdx.x;
  const float* row = pre + (size_t)o * NPOS;
  float s = 0.f, q = 0.f;
  for (int p = threadIdx.x; p < NPOS; p += 256) {
    float v = row[p];
    s += v;
    q += v * v;
  }
  ls[threadIdx.x] = s;
  lq[threadIdx.x] = q;
  __syncthreads();
  for (int st = 128; st > 0; st >>= 1) {
    if (threadIdx.x < st) {
      ls[threadIdx.x] += ls[threadIdx.x + st];
      lq[threadIdx.x] += lq[threadIdx.x + st];
    }
    __syncthreads();
  }
  if (threadIdx.x == 0) { bsum[o] = ls[0]; bsq[o] = lq[0]; }
}

// ---------------------------------------------------------------------------
// BN (batch stats) + ReLU; writes into [B, Ctot, T, H, W] at channel offset.
// ---------------------------------------------------------------------------
__global__ __launch_bounds__(256)
void bnrelu(const float* __restrict__ pre, const float* __restrict__ bsum,
            const float* __restrict__ bsq, const float* __restrict__ gamma,
            const float* __restrict__ beta, float* __restrict__ dst,
            int Ctot, int choff) {
  const int idx = blockIdx.x * 256 + threadIdx.x;   // grid = Cout*NPOS/256
  const int o = idx / NPOS, p = idx - o * NPOS;
  const int b = p / THW, s = p - b * THW;
  const float inv = 1.f / (float)NPOS;
  const float mean = bsum[o] * inv;
  const float var  = bsq[o] * inv - mean * mean;
  float v = (pre[idx] - mean) * rsqrtf(var + 1e-5f);
  v = gamma[o] * v + beta[o];
  dst[((size_t)b * Ctot + choff + o) * THW + s] = fmaxf(v, 0.f);
}

// ---------------------------------------------------------------------------
// 3x3x3 max pool, pad 1, stride 1 (per b,c plane).
// ---------------------------------------------------------------------------
__global__ __launch_bounds__(256)
void maxpool3(const float* __restrict__ x, float* __restrict__ xm) {
  const int idx = blockIdx.x * 256 + threadIdx.x;   // grid = 832*NPOS/256
  const int bc = idx / THW, s = idx - bc * THW;
  const int t = s / HWX, r = s - t * HWX, h = r / TWw, w = r - h * TWw;
  const float* base = x + (size_t)bc * THW;
  float m = -INFINITY;
  for (int dt = -1; dt <= 1; ++dt) {
    const int tt = t + dt;
    if (tt < 0 || tt >= TT) continue;
    for (int dh = -1; dh <= 1; ++dh) {
      const int hh = h + dh;
      if (hh < 0 || hh >= THh) continue;
      for (int dw = -1; dw <= 1; ++dw) {
        const int ww = w + dw;
        if (ww < 0 || ww >= TWw) continue;
        m = fmaxf(m, base[(tt * THh + hh) * TWw + ww]);
      }
    }
  }
  xm[idx] = m;
}

// ---------------------------------------------------------------------------
extern "C" void kernel_launch(void* const* d_in, const int* in_sizes, int n_in,
                              void* d_out, int out_size, void* d_ws, size_t ws_size,
                              hipStream_t stream) {
  (void)in_sizes; (void)n_in; (void)out_size; (void)ws_size;
  const float* x = (const float*)d_in[0];
  #define WOFFP(i) ((const float*)d_in[1 + 5 * (i)])
  #define BOFFP(i) ((const float*)d_in[2 + 5 * (i)])
  #define WP(i)    ((const float*)d_in[3 + 5 * (i)])
  #define GP(i)    ((const float*)d_in[4 + 5 * (i)])
  #define BTP(i)   ((const float*)d_in[5 + 5 * (i)])
  float* out = (float*)d_out;

  // Workspace partition (floats): total ~17.9M floats (~72 MB).
  float* ws  = (float*)d_ws;
  float* pre = ws;                                   // 320*NPOS
  float* xm  = pre + (size_t)320 * NPOS;             // 832*NPOS
  float* y1  = xm + (size_t)832 * NPOS;              // 160*NPOS
  float* y2  = y1 + (size_t)160 * NPOS;              // 32*NPOS
  float* off = y2 + (size_t)32 * NPOS;               // 81*NPOS
  float* bs  = off + (size_t)81 * NPOS;              // 320 (sum)
  float* bq  = bs + 320;                             // 320 (sumsq)

  const dim3 b128(128), b256(256);
  const int gOff1 = NPOS / 128;        // 98
  const int gOff3 = 81 * NPOS / 256;   // 3969
  const int gGemm = NPOS / 16;         // 784
  const int gPool = 832 * NPOS / 256;  // 40768

  // Branch 0: 832 -> 256, k=1  -> out channels [0,256)
  offc1<<<gOff1, b128, 0, stream>>>(x, WOFFP(0), BOFFP(0), off);
  deform_gemm<1, 0, 832, 256><<<gGemm, b128, 0, stream>>>(x, off, WP(0), pre);
  bnstats<<<256, b256, 0, stream>>>(pre, bs, bq);
  bnrelu<<<256 * NPOS / 256, b256, 0, stream>>>(pre, bs, bq, GP(0), BTP(0), out, 832, 0);

  // Branch 1: 832 -> 160 (k=1) -> 320 (k=3) -> out channels [256,576)
  offc1<<<gOff1, b128, 0, stream>>>(x, WOFFP(1), BOFFP(1), off);
  deform_gemm<1, 0, 832, 160><<<gGemm, b128, 0, stream>>>(x, off, WP(1), pre);
  bnstats<<<160, b256, 0, stream>>>(pre, bs, bq);
  bnrelu<<<160 * NPOS / 256, b256, 0, stream>>>(pre, bs, bq, GP(1), BTP(1), y1, 160, 0);
  offc3<<<gOff3, b256, 0, stream>>>(y1, WOFFP(2), BOFFP(2), off, 160);
  deform_gemm<27, 1, 160, 320><<<gGemm, b128, 0, stream>>>(y1, off, WP(2), pre);
  bnstats<<<320, b256, 0, stream>>>(pre, bs, bq);
  bnrelu<<<320 * NPOS / 256, b256, 0, stream>>>(pre, bs, bq, GP(2), BTP(2), out, 832, 256);

  // Branch 2: 832 -> 32 (k=1) -> 128 (k=3) -> out channels [576,704)
  offc1<<<gOff1, b128, 0, stream>>>(x, WOFFP(3), BOFFP(3), off);
  deform_gemm<1, 0, 832, 32><<<gGemm, b128, 0, stream>>>(x, off, WP(3), pre);
  bnstats<<<32, b256, 0, stream>>>(pre, bs, bq);
  bnrelu<<<32 * NPOS / 256, b256, 0, stream>>>(pre, bs, bq, GP(3), BTP(3), y2, 32, 0);
  offc3<<<gOff3, b256, 0, stream>>>(y2, WOFFP(4), BOFFP(4), off, 32);
  deform_gemm<27, 1, 32, 128><<<gGemm, b128, 0, stream>>>(y2, off, WP(4), pre);
  bnstats<<<128, b256, 0, stream>>>(pre, bs, bq);
  bnrelu<<<128 * NPOS / 256, b256, 0, stream>>>(pre, bs, bq, GP(4), BTP(4), out, 832, 576);

  // Branch 3: maxpool -> 832 -> 128 (k=1) -> out channels [704,832)
  maxpool3<<<gPool, b256, 0, stream>>>(x, xm);
  offc1<<<gOff1, b128, 0, stream>>>(xm, WOFFP(5), BOFFP(5), off);
  deform_gemm<1, 0, 832, 128><<<gGemm, b128, 0, stream>>>(xm, off, WP(5), pre);
  bnstats<<<128, b256, 0, stream>>>(pre, bs, bq);
  bnrelu<<<128 * NPOS / 256, b256, 0, stream>>>(pre, bs, bq, GP(5), BTP(5), out, 832, 704);

  #undef WOFFP
  #undef BOFFP
  #undef WP
  #undef GP
  #undef BTP
}
